// LocalAttentionBlock_27401891349134
// MI455X (gfx1250) — compile-verified, matrix operand order checked by reading, not run on hardware
//
#include <hip/hip_runtime.h>
#include <hip/hip_bf16.h>

// Problem constants (match reference)
#define B_      2
#define T_      2048
#define WIDTH_  1024
#define NH_     16
#define HD_     64
#define WINDOW_ 1024

typedef __attribute__((ext_vector_type(16))) __bf16    v16bf;
typedef __attribute__((ext_vector_type(8)))  __bf16    v8bf;
typedef __attribute__((ext_vector_type(8)))  float     v8f;
typedef __attribute__((ext_vector_type(4)))  unsigned  u32x4;
typedef __attribute__((ext_vector_type(8)))  unsigned  u32x8;

// ---------------------------------------------------------------------------
// Fragment loader for the CDNA5 16-bit A-matrix 16x32 layout (ISA 7.12.2):
// lane L (0..15) holds row M=L, K in {0..7, 16..23}; lane L+16 holds row M=L,
// K in {8..15, 24..31}.  Element e<8 -> K = e + 8*kh; e>=8 -> K = e+8 + 8*kh.
// Both halves are 8 contiguous bf16 = one 16-byte load (global b128 or
// ds_load_b128, depending on the pointer's address space).
// ---------------------------------------------------------------------------
__device__ __forceinline__ v16bf frag_from_row(const __bf16* row, int k0, int kh) {
  const v8bf c0 = *(const v8bf*)(row + k0 + 8 * kh);
  const v8bf c1 = *(const v8bf*)(row + k0 + 16 + 8 * kh);
  v16bf r;
#pragma unroll
  for (int i = 0; i < 8; ++i) { r[i] = c0[i]; r[i + 8] = c1[i]; }
  return r;
}

// ---------------------------------------------------------------------------
// Tensor Data Mover: 2D tile load, global -> LDS  (ISA ch.8, D# groups 0/1).
// 2D tensor: VADDR2/VADDR3 are NULL -> two-operand tensor_load_to_lds form.
// data_size = 2 bytes (bf16). All sizes/strides in elements.
// LDS byte offset = low 32 bits of the generic pointer (ISA flat->LDS rule).
// ---------------------------------------------------------------------------
__device__ __forceinline__ void tdm_load_2d_bf16(const __bf16* gtile,
                                                 const __bf16* ldsdst,
                                                 unsigned tensor_d0,
                                                 unsigned tensor_d1,
                                                 unsigned stride_d0,
                                                 unsigned tile_d0,
                                                 unsigned tile_d1) {
  const unsigned long long ga = (unsigned long long)(const void*)gtile;
  const unsigned lds_off = (unsigned)(unsigned long long)(const void*)ldsdst;

  u32x4 g0;
  g0[0] = 1u;                                     // count=1, no gather
  g0[1] = lds_off;                                // lds_addr (bytes)
  g0[2] = (unsigned)(ga & 0xFFFFFFFFu);           // global_addr[31:0]
  g0[3] = (unsigned)((ga >> 32) & 0x1FFFFFFu)     // global_addr[56:32]
          | (2u << 30);                           // type = 2 ("image")

  u32x8 g1;
  g1[0] = (1u << 16);                             // data_size=1 -> 2 bytes
  g1[1] = (tensor_d0 & 0xFFFFu) << 16;            // tensor_dim0[15:0]
  g1[2] = ((tensor_d0 >> 16) & 0xFFFFu)           // tensor_dim0[31:16]
          | ((tensor_d1 & 0xFFFFu) << 16);        // tensor_dim1[15:0]
  g1[3] = ((tensor_d1 >> 16) & 0xFFFFu)           // tensor_dim1[31:16]
          | ((tile_d0 & 0xFFFFu) << 16);          // tile_dim0
  g1[4] = (tile_d1 & 0xFFFFu);                    // tile_dim1 (tile_dim2 = 0)
  g1[5] = stride_d0;                              // tensor_dim0_stride[31:0]
  g1[6] = 0u;                                     // stride hi / dim1_stride lo
  g1[7] = 0u;                                     // dim1_stride hi (2D: unused)

  asm volatile("tensor_load_to_lds %0, %1" ::"s"(g0), "s"(g1) : "memory");
}

// ---------------------------------------------------------------------------
// f32 -> bf16 conversion
// ---------------------------------------------------------------------------
__global__ void f32_to_bf16_kernel(const float* __restrict__ src,
                                   __bf16* __restrict__ dst, int n) {
  int i = blockIdx.x * blockDim.x + threadIdx.x;
  if (i < n) dst[i] = (__bf16)src[i];
}

// ---------------------------------------------------------------------------
// C[M,N] = A[M,K] @ W[N,K]^T   (A, W bf16; f32 accumulate via WMMA)
// Block: 128 threads = 4 waves; block tile 64(M) x 64(N); wave strip 16x64.
// A and W tiles (64x32 each) staged in LDS by the Tensor Data Mover,
// double-buffered: wave 0 issues the next pair of descriptors, waits
// TENSORcnt<=2 (previous pair landed), block barrier, all waves compute
// 4 WMMAs from ds_load_b128 fragments.
// out_mode: 0 = bf16 row-major [M,N]
//           1 = f32 row-major [M,N] + bias[n]
//           2 = bf16 per-batch transposed: C[(m/T)*N + n][T] + (m%T)
// ---------------------------------------------------------------------------
__global__ void gemm_nt_wmma(const __bf16* __restrict__ A,
                             const __bf16* __restrict__ W,
                             const float* __restrict__ bias,
                             void* __restrict__ Cout,
                             int M, int N, int K, int out_mode) {
  __shared__ __align__(16) __bf16 Abuf[2][64 * 32];
  __shared__ __align__(16) __bf16 Bbuf[2][64 * 32];

  const int lane  = threadIdx.x & 31;
  const int wave  = threadIdx.x >> 5;
  const int mlane = lane & 15;
  const int kh    = lane >> 4;

  const int mb = blockIdx.y * 64;        // block row base
  const int n0 = blockIdx.x * 64;        // block col base
  const int m0 = mb + wave * 16;

  const int nk = K >> 5;                 // 32-deep k-steps

  if (wave == 0) {
    tdm_load_2d_bf16(A + (size_t)mb * K, Abuf[0], (unsigned)K, 64u,
                     (unsigned)K, 32u, 64u);
    tdm_load_2d_bf16(W + (size_t)n0 * K, Bbuf[0], (unsigned)K, 64u,
                     (unsigned)K, 32u, 64u);
  }

  v8f acc[4] = {};
  for (int ki = 0; ki < nk; ++ki) {
    if (wave == 0) {
      if (ki + 1 < nk) {
        const int k1 = (ki + 1) << 5;
        tdm_load_2d_bf16(A + (size_t)mb * K + k1, Abuf[(ki + 1) & 1],
                         (unsigned)K, 64u, (unsigned)K, 32u, 64u);
        tdm_load_2d_bf16(W + (size_t)n0 * K + k1, Bbuf[(ki + 1) & 1],
                         (unsigned)K, 64u, (unsigned)K, 32u, 64u);
        __builtin_amdgcn_s_wait_tensorcnt(2);  // pair ki complete
      } else {
        __builtin_amdgcn_s_wait_tensorcnt(0);  // final pair complete
      }
    }
    __syncthreads();  // LDS tile (ki & 1) visible to all waves

    const __bf16* At = Abuf[ki & 1];
    const __bf16* Bt = Bbuf[ki & 1];
    const v16bf af = frag_from_row(At + (wave * 16 + mlane) * 32, 0, kh);
#pragma unroll
    for (int j = 0; j < 4; ++j) {
      const v16bf bfr = frag_from_row(Bt + (16 * j + mlane) * 32, 0, kh);
      acc[j] = __builtin_amdgcn_wmma_f32_16x16x32_bf16(
          false, af, false, bfr, (short)0, acc[j], false, false);
    }
    __syncthreads();  // all waves done reading before buffer is reused
  }

  // C/D layout: lane n = lane&15, row m = v + 8*(lane>>4)
#pragma unroll
  for (int j = 0; j < 4; ++j) {
#pragma unroll
    for (int v = 0; v < 8; ++v) {
      const int m = m0 + v + 8 * kh;
      const int n = n0 + 16 * j + mlane;
      const float val = acc[j][v];
      if (out_mode == 1) {
        ((float*)Cout)[(size_t)m * N + n] = val + bias[n];
      } else if (out_mode == 2) {
        ((__bf16*)Cout)[((size_t)((m >> 11) * N + n)) * T_ + (m & (T_ - 1))] =
            (__bf16)val;
      } else {
        ((__bf16*)Cout)[(size_t)m * N + n] = (__bf16)val;
      }
    }
  }
}

// ---------------------------------------------------------------------------
// Sliding-window flash attention.
// One wave owns a 16-query tile of one (batch, head). Key loop over 32-wide
// s-tiles in [t0-WINDOW, t0+15]: 4 QK^T WMMAs -> mask -> online softmax
// (shfl-xor reductions within 16-lane halves, matching C/D layout) -> P(bf16)
// through LDS (layout swap D->A) -> 4 PV WMMAs into f32 accumulators.
//  q : bf16 [B,T,NH*HD]   k : bf16 [B,T,HD]   vt : bf16 [B,HD,T] (transposed)
//  o : bf16 [B,T,NH*HD]
// ---------------------------------------------------------------------------
__global__ void local_attn_wmma(const __bf16* __restrict__ qb,
                                const __bf16* __restrict__ kb,
                                const __bf16* __restrict__ vtb,
                                __bf16* __restrict__ ob) {
  __shared__ __align__(16) __bf16 Plds[4][16 * 32];

  const int lane  = threadIdx.x & 31;
  const int wave  = threadIdx.x >> 5;
  const int mlane = lane & 15;
  const int kh    = lane >> 4;

  const int NT   = T_ / 16;                       // 128 query tiles per head
  const int tile = blockIdx.x * 4 + wave;         // 0 .. B*NH*NT-1
  const int t0   = (tile & (NT - 1)) * 16;
  const int head = (tile >> 7) & (NH_ - 1);
  const int b    = tile >> 11;

  // Q fragments (K = HD = 64 -> two 16x32 fragments)
  const __bf16* qrow =
      qb + (size_t)(b * T_ + t0 + mlane) * WIDTH_ + head * HD_;
  const v16bf qf0 = frag_from_row(qrow, 0, kh);
  const v16bf qf1 = frag_from_row(qrow, 32, kh);

  float mstat[8], lsum[8];
  v8f acc[4] = {};
#pragma unroll
  for (int v = 0; v < 8; ++v) { mstat[v] = -1e30f; lsum[v] = 0.0f; }

  int sBeg = t0 - WINDOW_;
  if (sBeg < 0) sBeg = 0;
  sBeg &= ~31;
  const int sEnd = t0 + 16;
  const float scale = 0.125f;  // 1/sqrt(64)

  for (int s0 = sBeg; s0 < sEnd; s0 += 32) {
    // ---- logits: 16 x 32 tile via 4 WMMAs -------------------------------
    v8f lg[2] = {};
#pragma unroll
    for (int half = 0; half < 2; ++half) {
      const __bf16* krow =
          kb + (size_t)(b * T_ + s0 + 16 * half + mlane) * HD_;
      v16bf kf0 = frag_from_row(krow, 0, kh);
      lg[half] = __builtin_amdgcn_wmma_f32_16x16x32_bf16(
          false, qf0, false, kf0, (short)0, lg[half], false, false);
      v16bf kf1 = frag_from_row(krow, 32, kh);
      lg[half] = __builtin_amdgcn_wmma_f32_16x16x32_bf16(
          false, qf1, false, kf1, (short)0, lg[half], false, false);
    }

    // ---- mask + per-row tile max ----------------------------------------
    float pv[2][8];
    float vmax[8];
#pragma unroll
    for (int v = 0; v < 8; ++v) vmax[v] = -1e30f;
#pragma unroll
    for (int half = 0; half < 2; ++half) {
      const int scol = s0 + 16 * half + mlane;
#pragma unroll
      for (int v = 0; v < 8; ++v) {
        const int q = t0 + v + 8 * kh;
        const bool valid = (q >= scol) && (q <= scol + WINDOW_);
        const float x = valid ? lg[half][v] * scale : -1e30f;
        pv[half][v] = x;
        vmax[v] = fmaxf(vmax[v], x);
      }
    }
    // row reduction: rows m<8 live in lanes 0-15, rows 8-15 in lanes 16-31,
    // so xor masks 1..8 stay inside the owning 16-lane half.
#pragma unroll
    for (int v = 0; v < 8; ++v) {
#pragma unroll
      for (int off = 1; off < 16; off <<= 1)
        vmax[v] = fmaxf(vmax[v], __shfl_xor(vmax[v], off, 32));
    }

    // ---- online softmax state update ------------------------------------
    float alpha[8];
#pragma unroll
    for (int v = 0; v < 8; ++v) {
      const float mn = fmaxf(mstat[v], vmax[v]);
      alpha[v] = __expf(mstat[v] - mn);
      mstat[v] = mn;
      lsum[v] *= alpha[v];
    }
#pragma unroll
    for (int j = 0; j < 4; ++j)
#pragma unroll
      for (int v = 0; v < 8; ++v) acc[j][v] *= alpha[v];

    // ---- probabilities -> LDS (D-layout write, A-layout re-read) --------
#pragma unroll
    for (int half = 0; half < 2; ++half) {
      const int c = 16 * half + mlane;
#pragma unroll
      for (int v = 0; v < 8; ++v) {
        const int q = t0 + v + 8 * kh;
        const int scol = s0 + c;
        const bool valid = (q >= scol) && (q <= scol + WINDOW_);
        const float p = valid ? __expf(pv[half][v] - mstat[v]) : 0.0f;
        lsum[v] += p;
        Plds[wave][(v + 8 * kh) * 32 + c] = (__bf16)p;
      }
    }
    // same-wave LDS RAW: make the ds stores visible before ds_load_b128
    asm volatile("s_wait_dscnt 0" ::: "memory");

    // ---- PV: P(16x32) @ V(32x64) ----------------------------------------
    const __bf16* prow = &Plds[wave][mlane * 32];
    const v16bf pf = frag_from_row(prow, 0, kh);
#pragma unroll
    for (int j = 0; j < 4; ++j) {
      const __bf16* vrow = vtb + (size_t)(b * HD_ + 16 * j + mlane) * T_;
      v16bf vf = frag_from_row(vrow, s0, kh);
      acc[j] = __builtin_amdgcn_wmma_f32_16x16x32_bf16(
          false, pf, false, vf, (short)0, acc[j], false, false);
    }
  }

  // ---- normalize and store --------------------------------------------
#pragma unroll
  for (int v = 0; v < 8; ++v) {
#pragma unroll
    for (int off = 1; off < 16; off <<= 1)
      lsum[v] += __shfl_xor(lsum[v], off, 32);
  }
#pragma unroll
  for (int j = 0; j < 4; ++j) {
#pragma unroll
    for (int v = 0; v < 8; ++v) {
      const float o = acc[j][v] / lsum[v];
      const int m = t0 + v + 8 * kh;
      ob[(size_t)(b * T_ + m) * WIDTH_ + head * HD_ + 16 * j + mlane] =
          (__bf16)o;
    }
  }
}

// ---------------------------------------------------------------------------
// Launch: convert to bf16 once, then 4 TDM-staged WMMA GEMMs + flash attn.
// ---------------------------------------------------------------------------
extern "C" void kernel_launch(void* const* d_in, const int* in_sizes, int n_in,
                              void* d_out, int out_size, void* d_ws,
                              size_t ws_size, hipStream_t stream) {
  const float* x  = (const float*)d_in[0];
  // d_in[1] = segment_pos (unused: positions are arange in the reference)
  const float* Wq = (const float*)d_in[2];
  const float* Wk = (const float*)d_in[3];
  const float* Wv = (const float*)d_in[4];
  const float* Wf = (const float*)d_in[5];
  const float* bf = (const float*)d_in[6];
  float* out = (float*)d_out;

  char* ws = (char*)d_ws;
  size_t off = 0;
  auto alloc = [&](size_t bytes) -> void* {
    void* p = ws + off;
    off += (bytes + 255) & ~(size_t)255;
    return p;
  };
  __bf16* xb   = (__bf16*)alloc((size_t)B_ * T_ * WIDTH_ * 2);
  __bf16* Wqb  = (__bf16*)alloc((size_t)WIDTH_ * WIDTH_ * 2);
  __bf16* Wkb  = (__bf16*)alloc((size_t)HD_ * WIDTH_ * 2);
  __bf16* Wvb  = (__bf16*)alloc((size_t)HD_ * WIDTH_ * 2);
  __bf16* Wfb  = (__bf16*)alloc((size_t)WIDTH_ * WIDTH_ * 2);
  __bf16* qbuf = (__bf16*)alloc((size_t)B_ * T_ * WIDTH_ * 2);
  __bf16* kbuf = (__bf16*)alloc((size_t)B_ * T_ * HD_ * 2);
  __bf16* vtb  = (__bf16*)alloc((size_t)B_ * HD_ * T_ * 2);  // V^T per batch
  __bf16* ab   = (__bf16*)alloc((size_t)B_ * T_ * WIDTH_ * 2);

  int n;
  n = B_ * T_ * WIDTH_;
  f32_to_bf16_kernel<<<(n + 255) / 256, 256, 0, stream>>>(x, xb, n);
  n = WIDTH_ * WIDTH_;
  f32_to_bf16_kernel<<<(n + 255) / 256, 256, 0, stream>>>(Wq, Wqb, n);
  n = HD_ * WIDTH_;
  f32_to_bf16_kernel<<<(n + 255) / 256, 256, 0, stream>>>(Wk, Wkb, n);
  f32_to_bf16_kernel<<<(n + 255) / 256, 256, 0, stream>>>(Wv, Wvb, n);
  n = WIDTH_ * WIDTH_;
  f32_to_bf16_kernel<<<(n + 255) / 256, 256, 0, stream>>>(Wf, Wfb, n);

  const int M = B_ * T_;  // 4096
  // Q projection: [M,1024] = x @ Wq^T  (bf16 out)
  gemm_nt_wmma<<<dim3(WIDTH_ / 64, M / 64), 128, 0, stream>>>(
      xb, Wqb, nullptr, qbuf, M, WIDTH_, WIDTH_, 0);
  // K projection: [M,64] (bf16 out, row-major)
  gemm_nt_wmma<<<dim3(1, M / 64), 128, 0, stream>>>(
      xb, Wkb, nullptr, kbuf, M, HD_, WIDTH_, 0);
  // V projection: [M,64] stored per-batch transposed [B,64,T]
  gemm_nt_wmma<<<dim3(1, M / 64), 128, 0, stream>>>(
      xb, Wvb, nullptr, vtb, M, HD_, WIDTH_, 2);
  // Attention: B*NH*(T/16) = 4096 wave-tiles, 4 waves/block
  local_attn_wmma<<<dim3((B_ * NH_ * (T_ / 16)) / 4), 128, 0, stream>>>(
      qbuf, kbuf, vtb, ab);
  // Output projection with bias: f32 straight into d_out
  gemm_nt_wmma<<<dim3(WIDTH_ / 64, M / 64), 128, 0, stream>>>(
      ab, Wfb, bf, out, M, WIDTH_, WIDTH_, 1);
}
